// BiDirectionalConcatBlock_59622736003618
// MI455X (gfx1250) — compile-verified, matrix operand order checked by reading, not run on hardware
//
#include <hip/hip_runtime.h>
#include <math.h>

// ---------------- problem constants ----------------
#define BB 8
#define LL 1024
#define DIMM 1024
#define D_MODEL 512
#define D_INNER 1024
#define D_STATE 16
#define D_CONV 4
#define DT_RANK 32
#define MROWS (BB * LL)          // 8192

// ---------------- WMMA types ----------------
typedef __attribute__((ext_vector_type(16))) __bf16 v16bf;
typedef __attribute__((ext_vector_type(8)))  float  v8f;

union Frag { uint4 q[2]; v16bf v; };

__device__ __forceinline__ unsigned short f2bf(float f) {
  unsigned int u = __float_as_uint(f);
  unsigned int r = u + 0x7FFFu + ((u >> 16) & 1u);
  return (unsigned short)(r >> 16);
}

__device__ __forceinline__ float silu_f(float v) {
  return v / (1.0f + __expf(-v));
}

__device__ __forceinline__ unsigned lds_addr_u32(const void* p) {
  // __shared__ pointers: low 32 bits of the generic address are the LDS offset
  return (unsigned)(unsigned long long)(uintptr_t)p;
}

// ---------------- LayerNorm -> bf16 (normal + reversed) ----------------
__global__ __launch_bounds__(256)
void ln_kernel(const float* __restrict__ x, const float* __restrict__ g,
               const float* __restrict__ b,
               unsigned short* __restrict__ hbf,
               unsigned short* __restrict__ hbf_rev) {
  int row = blockIdx.x;                      // b*L + t
  const float* xr = x + (size_t)row * DIMM;
  int tid = threadIdx.x, lane = tid & 31, wv = tid >> 5;
  __shared__ float rs[8], rs2[8];

  float v[4], s = 0.f, s2 = 0.f;
#pragma unroll
  for (int i = 0; i < 4; ++i) {
    v[i] = xr[tid + i * 256];
    s += v[i]; s2 += v[i] * v[i];
  }
#pragma unroll
  for (int off = 16; off; off >>= 1) {
    s  += __shfl_xor(s,  off);
    s2 += __shfl_xor(s2, off);
  }
  if (lane == 0) { rs[wv] = s; rs2[wv] = s2; }
  __syncthreads();
  if (tid == 0) {
    float S = 0.f, S2 = 0.f;
    for (int i = 0; i < 8; ++i) { S += rs[i]; S2 += rs2[i]; }
    rs[0] = S; rs2[0] = S2;
  }
  __syncthreads();
  float mu  = rs[0] * (1.0f / DIMM);
  float var = rs2[0] * (1.0f / DIMM) - mu * mu;
  float rstd = rsqrtf(var + 1e-5f);

  int bb = row / LL, t = row % LL;
  size_t rrev = (size_t)bb * LL + (LL - 1 - t);
#pragma unroll
  for (int i = 0; i < 4; ++i) {
    int c = tid + i * 256;
    float hv = (v[i] - mu) * rstd * g[c] + b[c];
    unsigned short h = f2bf(hv);
    hbf[(size_t)row * DIMM + c] = h;
    hbf_rev[rrev * DIMM + c]    = h;
  }
}

// ---------------- fp32 (K x N) -> bf16 transposed (N x K) ----------------
__global__ __launch_bounds__(256)
void cvtT_kernel(const float* __restrict__ W, unsigned short* __restrict__ WT,
                 int K, int N) {
  size_t i = (size_t)blockIdx.x * 256 + threadIdx.x;
  if (i >= (size_t)K * N) return;
  int nn = (int)(i % N);
  int kk = (int)(i / N);
  WT[(size_t)nn * K + kk] = f2bf(W[i]);
}

// ---------------- tiled bf16 WMMA GEMM: C = A(MxK) * B(KxN) ----------------
// A: row-major bf16 (M multiple of TM assumed in-bounds -> async staged).
// Bt: B transposed (N x K) row-major bf16; async staged when tile fully
// in-bounds, guarded VGPR path otherwise (GEMM2, N=64).
// mode 0: plain store   mode 1: store + bf16 aux for cols<DT_RANK
// mode 2: +bias then softplus
#define TM 256
#define TN 128
#define TK 32
#define LSTR 40   // shorts per LDS row (padded: 80B, 16B-aligned chunks)

__global__ __launch_bounds__(256)
void gemm_bf16_kernel(const unsigned short* __restrict__ A,
                      const unsigned short* __restrict__ Bt,
                      float* __restrict__ C, int ldc, int col_off,
                      int M, int N, int K,
                      int mode, const float* __restrict__ bias,
                      unsigned short* __restrict__ aux) {
  __shared__ unsigned short sA[2][TM * LSTR];
  __shared__ unsigned short sB[2][TN * LSTR];

  int tid  = threadIdx.x;
  int lane = tid & 31, wave = tid >> 5;
  int wm = (wave & 3) * 64;       // wave M offset within block tile (4 waves)
  int wn = (wave >> 2) * 64;      // wave N offset within block tile (2 waves)
  int m0 = blockIdx.y * TM;
  int n0 = blockIdx.x * TN;
  int r  = lane & 15;
  int hi = lane >> 4;

  v8f acc[4][4];
  v8f zerov = {0.f, 0.f, 0.f, 0.f, 0.f, 0.f, 0.f, 0.f};
#pragma unroll
  for (int i = 0; i < 4; ++i)
#pragma unroll
    for (int j = 0; j < 4; ++j) acc[i][j] = zerov;

  // staging assignments
  int arow  = tid;                 // A: one full 32-short row per thread
  int brow  = tid >> 1;            // B: half row per thread
  int bhalf = (tid & 1) * 16;      // shorts
  bool bFull = (n0 + TN <= N);
  const unsigned short* gA = A  + (size_t)(m0 + arow) * K;
  const unsigned short* gB = Bt + (size_t)(n0 + brow) * K + bhalf;
  uint4 z4; z4.x = z4.y = z4.z = z4.w = 0u;

  auto stage = [&](int k0, int buf) {
    // A tile: 256 rows x 64B, async global->LDS (always in-bounds, M%TM==0)
    unsigned la = lds_addr_u32(&sA[buf][arow * LSTR]);
    unsigned long long sa = (unsigned long long)(uintptr_t)(gA + k0);
    asm volatile(
        "global_load_async_to_lds_b128 %0, %1, off\n\t"
        "global_load_async_to_lds_b128 %0, %1, off offset:16\n\t"
        "global_load_async_to_lds_b128 %0, %1, off offset:32\n\t"
        "global_load_async_to_lds_b128 %0, %1, off offset:48"
        :: "v"(la), "v"(sa) : "memory");
    if (bFull) {
      unsigned lb = lds_addr_u32(&sB[buf][brow * LSTR + bhalf]);
      unsigned long long sb = (unsigned long long)(uintptr_t)(gB + k0);
      asm volatile(
          "global_load_async_to_lds_b128 %0, %1, off\n\t"
          "global_load_async_to_lds_b128 %0, %1, off offset:16"
          :: "v"(lb), "v"(sb) : "memory");
    } else {
      uint4 b0 = z4, b1 = z4;
      if (n0 + brow < N) {
        const unsigned short* src = gB + k0;
        b0 = *(const uint4*)(src);
        b1 = *(const uint4*)(src + 8);
      }
      *(uint4*)(&sB[buf][brow * LSTR + bhalf])     = b0;
      *(uint4*)(&sB[buf][brow * LSTR + bhalf + 8]) = b1;
    }
  };

  int kt = K / TK;
  stage(0, 0);
  asm volatile("s_wait_asynccnt 0x0" ::: "memory");
  __syncthreads();

  for (int ki = 0; ki < kt; ++ki) {
    int buf = ki & 1;
    if (ki + 1 < kt) stage((ki + 1) * TK, buf ^ 1);

    Frag fa[4], fb[4];
#pragma unroll
    for (int mi = 0; mi < 4; ++mi) {
      const unsigned short* p = &sA[buf][(wm + mi * 16 + r) * LSTR + hi * 8];
      fa[mi].q[0] = *(const uint4*)(p);
      fa[mi].q[1] = *(const uint4*)(p + 16);
    }
#pragma unroll
    for (int ni = 0; ni < 4; ++ni) {
      const unsigned short* p = &sB[buf][(wn + ni * 16 + r) * LSTR + hi * 8];
      fb[ni].q[0] = *(const uint4*)(p);
      fb[ni].q[1] = *(const uint4*)(p + 16);
    }
#pragma unroll
    for (int mi = 0; mi < 4; ++mi)
#pragma unroll
      for (int ni = 0; ni < 4; ++ni)
        acc[mi][ni] = __builtin_amdgcn_wmma_f32_16x16x32_bf16(
            false, fa[mi].v, false, fb[ni].v, (short)0, acc[mi][ni],
            false, false);

    asm volatile("s_wait_asynccnt 0x0" ::: "memory");
    __syncthreads();
  }

  // ---- epilogue ----
#pragma unroll
  for (int mi = 0; mi < 4; ++mi) {
#pragma unroll
    for (int ni = 0; ni < 4; ++ni) {
      int gn = n0 + wn + ni * 16 + r;
      if (gn >= N) continue;
      int gm_base = m0 + wm + mi * 16 + hi * 8;
#pragma unroll
      for (int i = 0; i < 8; ++i) {
        int gm = gm_base + i;
        if (gm >= M) continue;
        float v = acc[mi][ni][i];
        if (mode == 2) {
          v += bias[gn];
          v = (v > 20.0f) ? v : log1pf(__expf(v));   // softplus
        }
        C[(size_t)gm * ldc + col_off + gn] = v;
        if (mode == 1 && gn < DT_RANK)
          aux[(size_t)gm * DT_RANK + gn] = f2bf(v);
      }
    }
  }
}

// ---------------- depthwise conv(4) + SiLU ----------------
__global__ __launch_bounds__(256)
void conv_silu_kernel(const float* __restrict__ xz,
                      const float* __restrict__ Wconv,
                      const float* __restrict__ bconv,
                      float* __restrict__ xc32,
                      unsigned short* __restrict__ xcbf) {
  size_t idx = (size_t)blockIdx.x * 256 + threadIdx.x;   // over B*L*D_INNER
  if (idx >= (size_t)MROWS * D_INNER) return;
  int c = (int)(idx & (D_INNER - 1));
  size_t row = idx >> 10;          // b*L + t
  int t = (int)(row % LL);
  float acc = bconv[c];
#pragma unroll
  for (int k = 0; k < D_CONV; ++k) {
    int tt = t + k - (D_CONV - 1);
    if (tt >= 0)
      acc += xz[(row + (size_t)(k - (D_CONV - 1))) * (2 * D_INNER) + c] *
             Wconv[c * D_CONV + k];
  }
  float s = silu_f(acc);
  xc32[idx] = s;
  xcbf[idx] = f2bf(s);
}

// ---------------- selective scan (one state element per lane) ----------------
__global__ __launch_bounds__(256)
void scan_kernel(const float* __restrict__ dt32,
                 const float* __restrict__ xc32,
                 const float* __restrict__ xz,
                 const float* __restrict__ dbc,
                 const float* __restrict__ Alog,
                 const float* __restrict__ Dp,
                 unsigned short* __restrict__ ybf) {
  int tid = threadIdx.x;
  int n   = tid & 15;                       // state index
  int chl = tid >> 4;                       // 0..15 channel within block
  int blk = blockIdx.x;                     // 0..511
  int bb  = blk >> 6;                       // 64 blocks per batch
  int d   = ((blk & 63) << 4) + chl;        // channel 0..1023

  float aL2 = -__expf(Alog[d * D_STATE + n]) * 1.44269504f;  // A*log2(e)
  float dpv = Dp[d];
  float h = 0.f;
  size_t rowbase = (size_t)bb * LL;

  for (int t = 0; t < LL; ++t) {
    size_t row = rowbase + t;
    float dtv = dt32[row * D_INNER + d];
    float xv  = xc32[row * D_INNER + d];
    float Bn  = dbc[row * 64 + DT_RANK + n];
    float Cn  = dbc[row * 64 + DT_RANK + D_STATE + n];
    h = exp2f(dtv * aL2) * h + dtv * Bn * xv;
    float p = h * Cn;
    p += __shfl_xor(p, 1);
    p += __shfl_xor(p, 2);
    p += __shfl_xor(p, 4);
    p += __shfl_xor(p, 8);
    if (n == 0) {
      float zv = xz[row * (2 * D_INNER) + D_INNER + d];
      float y = (p + dpv * xv) * silu_f(zv);
      ybf[row * D_INNER + d] = f2bf(y);
    }
  }
}

// ---------------- residual + exact GELU, in-place on d_out ----------------
__global__ __launch_bounds__(256)
void gelu_res_kernel(const float* __restrict__ x, float* __restrict__ out,
                     size_t n) {
  size_t i = (size_t)blockIdx.x * 256 + threadIdx.x;
  if (i >= n) return;
  float v = out[i] + x[i];
  out[i] = 0.5f * v * (1.0f + erff(v * 0.70710678118654752f));
}

// ---------------- orchestration ----------------
extern "C" void kernel_launch(void* const* d_in, const int* in_sizes, int n_in,
                              void* d_out, int out_size, void* d_ws,
                              size_t ws_size, hipStream_t stream) {
  (void)in_sizes; (void)n_in; (void)out_size; (void)ws_size;
  const float* x    = (const float*)d_in[0];
  const float* ln_g = (const float*)d_in[1];
  const float* ln_b = (const float*)d_in[2];

  // workspace carve (256B aligned)
  char* w = (char*)d_ws;
  auto carve = [&](size_t bytes) -> void* {
    void* p = (void*)w;
    w += (bytes + 255) & ~(size_t)255;
    return p;
  };
  unsigned short* hbf     = (unsigned short*)carve((size_t)MROWS * DIMM * 2);
  unsigned short* hbf_rev = (unsigned short*)carve((size_t)MROWS * DIMM * 2);
  unsigned short* WinT    = (unsigned short*)carve((size_t)2 * D_INNER * DIMM * 2);
  unsigned short* WxT     = (unsigned short*)carve((size_t)(DT_RANK + 2 * D_STATE) * D_INNER * 2);
  unsigned short* WdtT    = (unsigned short*)carve((size_t)D_INNER * DT_RANK * 2);
  unsigned short* WoutT   = (unsigned short*)carve((size_t)D_MODEL * D_INNER * 2);
  float*          xzbuf   = (float*)carve((size_t)MROWS * 2 * D_INNER * 4);
  float*          xc32    = (float*)carve((size_t)MROWS * D_INNER * 4);
  unsigned short* xcbf    = (unsigned short*)carve((size_t)MROWS * D_INNER * 2);
  float*          dbc     = (float*)carve((size_t)MROWS * 64 * 4);
  unsigned short* dtin    = (unsigned short*)carve((size_t)MROWS * DT_RANK * 2);
  float*          dt32    = (float*)carve((size_t)MROWS * D_INNER * 4);
  unsigned short* ybf     = (unsigned short*)carve((size_t)MROWS * D_INNER * 2);

  // 1) LayerNorm -> bf16 (normal + reversed)
  ln_kernel<<<MROWS, 256, 0, stream>>>(x, ln_g, ln_b, hbf, hbf_rev);

  for (int dir = 0; dir < 2; ++dir) {
    int base = 3 + dir * 9;
    const float* Win   = (const float*)d_in[base + 0];
    const float* Wconv = (const float*)d_in[base + 1];
    const float* bconv = (const float*)d_in[base + 2];
    const float* Wx    = (const float*)d_in[base + 3];
    const float* Wdt   = (const float*)d_in[base + 4];
    const float* bdt   = (const float*)d_in[base + 5];
    const float* Alog  = (const float*)d_in[base + 6];
    const float* Dp    = (const float*)d_in[base + 7];
    const float* Wout  = (const float*)d_in[base + 8];
    const unsigned short* hin = (dir == 0) ? hbf : hbf_rev;

    // weight conversions (fp32 KxN -> bf16 NxK)
    {
      size_t e;
      e = (size_t)DIMM * 2 * D_INNER;
      cvtT_kernel<<<(unsigned)((e + 255) / 256), 256, 0, stream>>>(Win, WinT, DIMM, 2 * D_INNER);
      e = (size_t)D_INNER * (DT_RANK + 2 * D_STATE);
      cvtT_kernel<<<(unsigned)((e + 255) / 256), 256, 0, stream>>>(Wx, WxT, D_INNER, DT_RANK + 2 * D_STATE);
      e = (size_t)DT_RANK * D_INNER;
      cvtT_kernel<<<(unsigned)((e + 255) / 256), 256, 0, stream>>>(Wdt, WdtT, DT_RANK, D_INNER);
      e = (size_t)D_INNER * D_MODEL;
      cvtT_kernel<<<(unsigned)((e + 255) / 256), 256, 0, stream>>>(Wout, WoutT, D_INNER, D_MODEL);
    }

    // GEMM1: xz = h @ Win   (8192 x 2048, K=1024)
    {
      dim3 g((2 * D_INNER + TN - 1) / TN, (MROWS + TM - 1) / TM);
      gemm_bf16_kernel<<<g, 256, 0, stream>>>(hin, WinT, xzbuf, 2 * D_INNER, 0,
                                              MROWS, 2 * D_INNER, DIMM,
                                              0, nullptr, nullptr);
    }

    // conv + silu
    {
      size_t e = (size_t)MROWS * D_INNER;
      conv_silu_kernel<<<(unsigned)((e + 255) / 256), 256, 0, stream>>>(
          xzbuf, Wconv, bconv, xc32, xcbf);
    }

    // GEMM2: dbc = xc @ Wx  (8192 x 64, K=1024); also emit bf16 of first 32 cols
    {
      dim3 g((64 + TN - 1) / TN, (MROWS + TM - 1) / TM);
      gemm_bf16_kernel<<<g, 256, 0, stream>>>(xcbf, WxT, dbc, 64, 0,
                                              MROWS, DT_RANK + 2 * D_STATE, DIMM,
                                              1, nullptr, dtin);
    }

    // GEMM3: dt = softplus(dbc[:, :32] @ Wdt + bdt)  (8192 x 1024, K=32)
    {
      dim3 g((D_INNER + TN - 1) / TN, (MROWS + TM - 1) / TM);
      gemm_bf16_kernel<<<g, 256, 0, stream>>>(dtin, WdtT, dt32, D_INNER, 0,
                                              MROWS, D_INNER, DT_RANK,
                                              2, bdt, nullptr);
    }

    // selective scan (+gate, +Dp*x), emits bf16 y
    scan_kernel<<<512, 256, 0, stream>>>(dt32, xc32, xzbuf, dbc, Alog, Dp, ybf);

    // GEMM4: out_half = y @ Wout  -> write into d_out at column dir*512
    {
      dim3 g((D_MODEL + TN - 1) / TN, (MROWS + TM - 1) / TM);
      gemm_bf16_kernel<<<g, 256, 0, stream>>>(ybf, WoutT, (float*)d_out, DIMM,
                                              dir * D_MODEL,
                                              MROWS, D_MODEL, DIMM,
                                              0, nullptr, nullptr);
    }
  }

  // residual + exact GELU, in-place on d_out
  {
    size_t e = (size_t)MROWS * DIMM;
    gelu_res_kernel<<<(unsigned)((e + 255) / 256), 256, 0, stream>>>(
        x, (float*)d_out, e);
  }
}
